// TokenPruner_8641474200503
// MI455X (gfx1250) — matrix operand order repaired and checked by reading, not compiled, not run on hardware
//
#include <hip/hip_runtime.h>
#include <hip/hip_bf16.h>

#define Bdim   16
#define Ndim   8192
#define Cdim   256
#define HIDdim 128
#define NKEEP  5734

typedef __attribute__((ext_vector_type(16))) _Float16 v16h;
typedef __attribute__((ext_vector_type(8)))  float    v8f;

// gfx1250 has a hardware tanh (v_tanh_f32) — branch-free GELU for ranking.
__device__ __forceinline__ float fast_tanh(float x) {
#if __has_builtin(__builtin_amdgcn_tanhf)
  return __builtin_amdgcn_tanhf(x);
#elif __has_builtin(__builtin_amdgcn_tanh_f32)
  return __builtin_amdgcn_tanh_f32(x);
#else
  return tanhf(x);
#endif
}

__device__ __forceinline__ float gelu_tanh(float x) {
  const float z = 0.7978845608028654f * (x + 0.044715f * x * x * x);
  return 0.5f * x * (1.0f + fast_tanh(z));
}

// ---------------------------------------------------------------------------
// Kernel 1: fused importance scores = (GELU(tokens@W1 + b1) @ W2 + b2)
// One wave32 computes a 32-token (2 x 16-row) tile with
// v_wmma_f32_16x16x32_f16; each LDS B fragment feeds two WMMAs.
// W1 is staged in LDS pre-swizzled into WMMA B-fragment layout.
// ---------------------------------------------------------------------------
__global__ __launch_bounds__(256) void score_kernel(
    const float* __restrict__ tokens, const float* __restrict__ W1,
    const float* __restrict__ b1, const float* __restrict__ W2,
    const float* __restrict__ b2, float* __restrict__ scores)
{
  extern __shared__ _Float16 ldsW[]; // 8 kf * 8 nf * 32 lanes * 16 halves = 64KB
  const int t = threadIdx.x;

  // Cooperative load of W1 (f32 -> f16) into B-fragment layout:
  // fragment (kf,nf), lane L holds column n = nf*16 + (L&15),
  // halves e -> K = kf*32 + (L>=16)*16 + e  (contiguous 32B per lane).
  for (int idx = t; idx < 8 * 8 * 32 * 16; idx += 256) {
    const int e    = idx & 15;
    const int lane = (idx >> 4) & 31;
    const int nf   = (idx >> 9) & 7;
    const int kf   = idx >> 12;
    const int K    = kf * 32 + ((lane >> 4) << 4) + e;
    const int Ncol = nf * 16 + (lane & 15);
    ldsW[idx] = (_Float16)W1[K * HIDdim + Ncol];
  }

  const int wave = t >> 5;
  const int lane = t & 31;
  const int col  = lane & 15;
  const int tile = blockIdx.x * 8 + wave;   // global 32-token tile id
  const int b    = tile / (Ndim / 32);
  const int trow = tile % (Ndim / 32);

  // Hoist per-column bias / W2 scalars out of the matrix loop.
  float biasv[8], w2v[8];
#pragma unroll
  for (int nf = 0; nf < 8; ++nf) {
    biasv[nf] = b1[nf * 16 + col];
    w2v[nf]   = W2[nf * 16 + col];
  }
  const float bb = b2[0];

  // A-fragment loads (16x32 f16 per kf): lane L row M = L&15,
  // K chunk base kb = (L>=16)*16, halves 0..15 -> K kb+0..15
  const float* arow0 =
      tokens + ((size_t)b * Ndim + (size_t)trow * 32 + (lane & 15)) * Cdim;
  const float* arow1 = arow0 + (size_t)16 * Cdim;
  const int kb = (lane >> 4) * 16;

  v16h a0[8], a1[8];
#pragma unroll
  for (int kf = 0; kf < 8; ++kf) {
    {
      const float* pa = arow0 + kf * 32 + kb;
      const float4 x0 = ((const float4*)pa)[0];
      const float4 x1 = ((const float4*)pa)[1];
      const float4 y0 = ((const float4*)(pa + 8))[0];
      const float4 y1 = ((const float4*)(pa + 8))[1];
      v16h av;
      av[0]  = (_Float16)x0.x; av[1]  = (_Float16)x0.y;
      av[2]  = (_Float16)x0.z; av[3]  = (_Float16)x0.w;
      av[4]  = (_Float16)x1.x; av[5]  = (_Float16)x1.y;
      av[6]  = (_Float16)x1.z; av[7]  = (_Float16)x1.w;
      av[8]  = (_Float16)y0.x; av[9]  = (_Float16)y0.y;
      av[10] = (_Float16)y0.z; av[11] = (_Float16)y0.w;
      av[12] = (_Float16)y1.x; av[13] = (_Float16)y1.y;
      av[14] = (_Float16)y1.z; av[15] = (_Float16)y1.w;
      a0[kf] = av;
    }
    {
      const float* pa = arow1 + kf * 32 + kb;
      const float4 x0 = ((const float4*)pa)[0];
      const float4 x1 = ((const float4*)pa)[1];
      const float4 y0 = ((const float4*)(pa + 8))[0];
      const float4 y1 = ((const float4*)(pa + 8))[1];
      v16h av;
      av[0]  = (_Float16)x0.x; av[1]  = (_Float16)x0.y;
      av[2]  = (_Float16)x0.z; av[3]  = (_Float16)x0.w;
      av[4]  = (_Float16)x1.x; av[5]  = (_Float16)x1.y;
      av[6]  = (_Float16)x1.z; av[7]  = (_Float16)x1.w;
      av[8]  = (_Float16)y0.x; av[9]  = (_Float16)y0.y;
      av[10] = (_Float16)y0.z; av[11] = (_Float16)y0.w;
      av[12] = (_Float16)y1.x; av[13] = (_Float16)y1.y;
      av[14] = (_Float16)y1.z; av[15] = (_Float16)y1.w;
      a1[kf] = av;
    }
  }

  __syncthreads();   // W1 staging complete

  float p0[8], p1[8];
#pragma unroll
  for (int m = 0; m < 8; ++m) { p0[m] = 0.0f; p1[m] = 0.0f; }

#pragma unroll
  for (int nf = 0; nf < 8; ++nf) {
    v8f c0, c1;
#pragma unroll
    for (int m = 0; m < 8; ++m) { c0[m] = biasv[nf]; c1[m] = biasv[nf]; }
    // Each B fragment feeds two independent WMMA chains (tile0 / tile1).
#pragma unroll
    for (int kf = 0; kf < 8; ++kf) {
      const v16h bf = *((const v16h*)&ldsW[((kf * 8 + nf) * 32 + lane) * 16]);
      c0 = __builtin_amdgcn_wmma_f32_16x16x32_f16(
          false, a0[kf], false, bf, (short)0, c0, false, false);
      c1 = __builtin_amdgcn_wmma_f32_16x16x32_f16(
          false, a1[kf], false, bf, (short)0, c1, false, false);
    }
    // fused GELU + W2 column dot (partial per lane), branch-free
#pragma unroll
    for (int m = 0; m < 8; ++m) {
      p0[m] += gelu_tanh(c0[m]) * w2v[nf];
      p1[m] += gelu_tanh(c1[m]) * w2v[nf];
    }
  }

  // reduce partial sums across the 16 lanes of each half (xor bits 0..3)
#pragma unroll
  for (int off = 1; off < 16; off <<= 1) {
#pragma unroll
    for (int m = 0; m < 8; ++m) {
      p0[m] += __shfl_xor(p0[m], off, 32);
      p1[m] += __shfl_xor(p1[m], off, 32);
    }
  }

  if ((lane & 15) == 0) {
    const int rbase = (lane >> 4) * 8; // lanes<16: rows 0..7, lanes>=16: 8..15
    float* sp = scores + (size_t)b * Ndim + (size_t)trow * 32 + rbase;
#pragma unroll
    for (int m = 0; m < 8; ++m) {
      sp[m]      = p0[m] + bb;
      sp[16 + m] = p1[m] + bb;
    }
  }
}

// ---------------------------------------------------------------------------
// Block-wide (1024 thr) exclusive scan helper (wave32 shfl + LDS wave totals)
// ---------------------------------------------------------------------------
__device__ __forceinline__ unsigned block_exscan_1024(unsigned v,
                                                      unsigned* wavesum)
{
  const int lane = threadIdx.x & 31;
  const int wid  = threadIdx.x >> 5;
  unsigned  x    = v;
#pragma unroll
  for (int off = 1; off < 32; off <<= 1) {
    const unsigned y = __shfl_up(x, off, 32);
    if (lane >= off) x += y;
  }
  __syncthreads();           // protect wavesum reuse across calls
  if (lane == 31) wavesum[wid] = x;
  __syncthreads();
  if (wid == 0) {
    unsigned w = wavesum[lane];
#pragma unroll
    for (int off = 1; off < 32; off <<= 1) {
      const unsigned y = __shfl_up(w, off, 32);
      if (lane >= off) w += y;
    }
    wavesum[lane] = w;       // inclusive wave totals
  }
  __syncthreads();
  const unsigned wbase = (wid == 0) ? 0u : wavesum[wid - 1];
  return wbase + x - v;      // exclusive
}

// ---------------------------------------------------------------------------
// Kernel 2: per-batch top-NKEEP selection via 4-pass radix select on
// order-preserving uint keys, ties broken by lowest index, output ascending.
// (softmax skipped: monotone, does not change top-k)
// ---------------------------------------------------------------------------
__global__ __launch_bounds__(1024) void select_kernel(
    const float* __restrict__ scores, int* __restrict__ idx_out)
{
  __shared__ unsigned      ukey[Ndim];     // 32KB
  __shared__ unsigned      hist[256];
  __shared__ unsigned      wavesum[32];
  __shared__ unsigned char masks[1024];
  __shared__ unsigned      s_prefix, s_remaining;

  const int    b = blockIdx.x;
  const int    t = threadIdx.x;
  const float* s = scores + (size_t)b * Ndim;

  for (int i = t; i < Ndim; i += 1024) {
    unsigned u = __float_as_uint(s[i]);
    u          = (u & 0x80000000u) ? ~u : (u | 0x80000000u); // order-preserving
    ukey[i]    = u;
  }
  if (t == 0) { s_prefix = 0u; s_remaining = NKEEP; }
  __syncthreads();

  // radix select, MSB byte first
  for (int p = 3; p >= 0; --p) {
    for (int i = t; i < 256; i += 1024) hist[i] = 0u;
    __syncthreads();
    const unsigned pref    = s_prefix;
    const unsigned shift   = (unsigned)p * 8u;
    const unsigned mask_hi = (p == 3) ? 0u : (0xFFFFFFFFu << (shift + 8u));
    for (int i = t; i < Ndim; i += 1024) {
      const unsigned u = ukey[i];
      if ((u & mask_hi) == (pref & mask_hi))
        atomicAdd(&hist[(u >> shift) & 255u], 1u);
    }
    __syncthreads();
    if (t == 0) {
      unsigned rem = s_remaining, cum = 0u;
      for (int bb = 255; bb >= 0; --bb) {
        cum += hist[bb];
        if (cum >= rem) {
          s_remaining = rem - (cum - hist[bb]); // #ties still needed
          s_prefix    = pref | ((unsigned)bb << shift);
          break;
        }
      }
    }
    __syncthreads();
  }

  const unsigned T    = s_prefix;    // threshold key (NKEEP-th largest)
  const unsigned need = s_remaining; // # of ==T elements to keep (lowest idx)

  // scan 1: exclusive rank among ==T elements (8 elems/thread)
  const int base  = t * 8;
  unsigned  eqcnt = 0u;
#pragma unroll
  for (int j = 0; j < 8; ++j) eqcnt += (ukey[base + j] == T) ? 1u : 0u;
  unsigned er = block_exscan_1024(eqcnt, wavesum);

  // keep flags
  unsigned m8 = 0u, kcnt = 0u;
#pragma unroll
  for (int j = 0; j < 8; ++j) {
    const unsigned u = ukey[base + j];
    bool keep = false;
    if (u > T) keep = true;
    else if (u == T) { if (er < need) keep = true; ++er; }
    if (keep) { m8 |= 1u << j; ++kcnt; }
  }
  masks[t] = (unsigned char)m8;

  // scan 2: output positions (ascending index order by construction)
  unsigned kpos = block_exscan_1024(kcnt, wavesum);
  const unsigned mm = masks[t];
#pragma unroll
  for (int j = 0; j < 8; ++j)
    if (mm & (1u << j)) idx_out[(size_t)b * NKEEP + (kpos++)] = base + j;
}

// ---------------------------------------------------------------------------
// Kernel 3: gather kept token rows (pure HBM bandwidth, float4 copies)
// ---------------------------------------------------------------------------
__global__ __launch_bounds__(64) void gather_kernel(
    const float* __restrict__ tokens, const int* __restrict__ idx,
    float* __restrict__ out)
{
  const int    row = blockIdx.x;         // b*NKEEP + k
  const int    b   = row / NKEEP;
  const int    tok = idx[row];
  const float4* src =
      (const float4*)(tokens + ((size_t)b * Ndim + (size_t)tok) * Cdim);
  float4* dst = (float4*)(out + (size_t)row * Cdim);
  dst[threadIdx.x] = src[threadIdx.x];   // 64 lanes x 16B = 256 floats
}

// ---------------------------------------------------------------------------
extern "C" void kernel_launch(void* const* d_in, const int* in_sizes, int n_in,
                              void* d_out, int out_size, void* d_ws,
                              size_t ws_size, hipStream_t stream)
{
  const float* tokens = (const float*)d_in[0];
  const float* W1     = (const float*)d_in[1];
  const float* b1     = (const float*)d_in[2];
  const float* W2     = (const float*)d_in[3];
  const float* b2     = (const float*)d_in[4];

  float* pruned  = (float*)d_out;                                  // [B,NKEEP,C]
  int*   idx_out = (int*)((float*)d_out + (size_t)Bdim * NKEEP * Cdim);
  float* scores  = (float*)d_ws;                                   // [B,N]

  // 1) scores via WMMA, 32 tokens/wave (64KB dynamic LDS for W1 fragments)
  hipLaunchKernelGGL(score_kernel, dim3(Bdim * Ndim / 256), dim3(256),
                     64 * 1024, stream, tokens, W1, b1, W2, b2, scores);
  // 2) per-batch radix top-k + ascending index compaction
  hipLaunchKernelGGL(select_kernel, dim3(Bdim), dim3(1024), 0, stream, scores,
                     idx_out);
  // 3) gather kept rows
  hipLaunchKernelGGL(gather_kernel, dim3(Bdim * NKEEP), dim3(64), 0, stream,
                     tokens, idx_out, pruned);
}
